// SelfAttention_23115513987118
// MI455X (gfx1250) — matrix operand order repaired and checked by reading, not compiled, not run on hardware
//
#include <hip/hip_runtime.h>
#include <cstdint>

typedef __attribute__((ext_vector_type(16))) __bf16 v16bf;
typedef __attribute__((ext_vector_type(8)))  float  v8f;

// ---- CDNA5 async global->LDS staging (guarded; fallback = sync copy) ------
#if defined(__has_builtin)
#if __has_builtin(__builtin_amdgcn_global_load_async_to_lds_b128)
#define USE_ASYNC_LDS 1
#endif
#endif

#ifdef USE_ASYNC_LDS
#if __has_builtin(__builtin_amdgcn_s_wait_asynccnt)
#define WAIT_ASYNC(N) __builtin_amdgcn_s_wait_asynccnt((short)(N))
#else
#define WAIT_ASYNC(N) asm volatile("s_wait_asynccnt %0" ::"i"(N) : "memory")
#endif
#else
#define WAIT_ASYNC(N) ((void)0)
#endif

__device__ __forceinline__ void cp16_to_lds(__bf16* lds_dst, const __bf16* gsrc) {
#ifdef USE_ASYNC_LDS
  typedef int v4i __attribute__((__vector_size__(16)));
  typedef __attribute__((address_space(1))) v4i* g4p;
  typedef __attribute__((address_space(3))) v4i* l4p;
  __builtin_amdgcn_global_load_async_to_lds_b128(
      (g4p)(uintptr_t)gsrc, (l4p)(uint32_t)(uintptr_t)lds_dst, 0, 0);
#else
  *(uint4*)lds_dst = *(const uint4*)gsrc;
#endif
}

__device__ __forceinline__ v8f wmma_bf16(v16bf a, v16bf b, v8f c) {
  return __builtin_amdgcn_wmma_f32_16x16x32_bf16(
      /*neg_a=*/false, a, /*neg_b=*/false, b,
      /*c_mod=*/(short)0, c, /*reuse_a=*/false, /*reuse_b=*/false);
}

// A fragment (16x32 bf16, MxK): lane m = lane&15; K = half*16+j
__device__ __forceinline__ v16bf frag_a(const __bf16* base, int ldk, int lane) {
  int half = lane >> 4, m = lane & 15;
  const __bf16* p = base + m * ldk + half * 16;
  v16bf r;
#pragma unroll
  for (int j = 0; j < 8; ++j) { r[j] = p[j]; r[j + 8] = p[8 + j]; }
  return r;
}

// B fragment (32x16 bf16, KxN), source stored as [N][K]: lane n = lane&15; K = half*16+j
__device__ __forceinline__ v16bf frag_b(const __bf16* base, int ldk, int lane) {
  int half = lane >> 4, n = lane & 15;
  const __bf16* p = base + n * ldk + half * 16;
  v16bf r;
#pragma unroll
  for (int j = 0; j < 16; ++j) r[j] = p[j];
  return r;
}

// ---------------------------------------------------------------------------
// 0) f32 -> bf16 weight conversion (once per launch; tiny vs HBM bandwidth)
// ---------------------------------------------------------------------------
__global__ __launch_bounds__(256) void cvt_kernel(const float* __restrict__ src,
                                                  __bf16* __restrict__ dst) {
  int base = blockIdx.x * 1024 + threadIdx.x;
#pragma unroll
  for (int i = 0; i < 4; ++i) dst[base + i * 256] = (__bf16)src[base + i * 256];
}

// ---------------------------------------------------------------------------
// 1) GroupNorm: x[b][512][1024] f32 -> xnT[b][n][c] bf16
// ---------------------------------------------------------------------------
__global__ __launch_bounds__(256) void gn_kernel(
    const float* __restrict__ x, const float* __restrict__ gamma,
    const float* __restrict__ beta, __bf16* __restrict__ xnT) {
  __shared__ float red1[256];
  __shared__ float red2[256];
  const int g = blockIdx.x, b = blockIdx.y, t = threadIdx.x;
  const size_t base = ((size_t)b * 512 + g * 64) * 1024;
  float s1 = 0.f, s2 = 0.f;
#pragma unroll 4
  for (int i = 0; i < 256; ++i) {
    float v = x[base + t + i * 256];
    s1 += v; s2 += v * v;
  }
  red1[t] = s1; red2[t] = s2;
  __syncthreads();
  for (int s = 128; s > 0; s >>= 1) {
    if (t < s) { red1[t] += red1[t + s]; red2[t] += red2[t + s]; }
    __syncthreads();
  }
  const float mean = red1[0] * (1.0f / 65536.0f);
  const float var  = red2[0] * (1.0f / 65536.0f) - mean * mean;
  const float rstd = rsqrtf(var + 1e-5f);
#pragma unroll 4
  for (int i = 0; i < 256; ++i) {
    int idx = t + i * 256;           // idx = c_in_g*1024 + n
    int cg = idx >> 10, n = idx & 1023;
    int co = g * 64 + cg;
    float v = x[base + idx];
    float y = (v - mean) * rstd * gamma[co] + beta[co];
    xnT[((size_t)b * 1024 + n) * 512 + co] = (__bf16)y;
  }
}

// ---------------------------------------------------------------------------
// Shared GEMM machinery: block tile 128(M) x 128(N), K=512 in 32-steps,
// double-buffered LDS tiles staged with async global->LDS copies.
// 8 waves = 4(M) x 2(N); wave tile 32x64; acc[2][4] of 16x16 f32.
// ---------------------------------------------------------------------------
__device__ __forceinline__ void stage_tile(__bf16* dst, const __bf16* src, int t) {
  int e0 = t * 8;
  cp16_to_lds(dst + e0, src + (e0 >> 5) * 512 + (e0 & 31));
  int e1 = e0 + 2048;
  cp16_to_lds(dst + e1, src + (e1 >> 5) * 512 + (e1 & 31));
}

__device__ __forceinline__ void gemm_mainloop(
    const __bf16* __restrict__ Asrc,   // [M][512] bf16, offset to block row
    const __bf16* __restrict__ Bsrc,   // [N][512] bf16, offset to block row
    __bf16* As,                        // LDS [2][128*32]
    __bf16* Bs,                        // LDS [2][128*32]
    v8f acc[2][4], int t, int wid, int lane) {
  const int wm = wid & 3, wn = wid >> 2;
  stage_tile(As, Asrc, t);
  stage_tile(Bs, Bsrc, t);
  for (int ks = 0; ks < 16; ++ks) {
    if (ks < 15) {
      int nb = (ks + 1) & 1;
      stage_tile(As + nb * 4096, Asrc + (ks + 1) * 32, t);
      stage_tile(Bs + nb * 4096, Bsrc + (ks + 1) * 32, t);
      WAIT_ASYNC(4);
    } else {
      WAIT_ASYNC(0);
    }
    __syncthreads();
    const __bf16* Ac = As + (ks & 1) * 4096;
    const __bf16* Bc = Bs + (ks & 1) * 4096;
    v16bf af[2], bfr[4];
#pragma unroll
    for (int mi = 0; mi < 2; ++mi) af[mi] = frag_a(Ac + (wm * 32 + mi * 16) * 32, 32, lane);
#pragma unroll
    for (int nj = 0; nj < 4; ++nj) bfr[nj] = frag_b(Bc + (wn * 64 + nj * 16) * 32, 32, lane);
#pragma unroll
    for (int mi = 0; mi < 2; ++mi)
#pragma unroll
      for (int nj = 0; nj < 4; ++nj) acc[mi][nj] = wmma_bf16(af[mi], bfr[nj], acc[mi][nj]);
    __syncthreads();
  }
}

// ---------------------------------------------------------------------------
// 2) QKV GEMM: D[o][n] = sum_c wq_bf[o][c] * xnT[n][c] + b_qkv[o]
//    Epilogue scatters Q,K -> [b][h][n][d], V -> [b][h][d][n] (bf16).
// ---------------------------------------------------------------------------
__global__ __launch_bounds__(256) void qkv_gemm(
    const __bf16* __restrict__ Wbf, const float* __restrict__ bias,
    const __bf16* __restrict__ xnT, __bf16* __restrict__ qT,
    __bf16* __restrict__ kT, __bf16* __restrict__ vv) {
  __shared__ __align__(16) __bf16 As[2 * 128 * 32];
  __shared__ __align__(16) __bf16 Bs[2 * 128 * 32];
  const int b = blockIdx.z;
  const int m0 = blockIdx.x * 128, n0 = blockIdx.y * 128;
  const int t = threadIdx.x, wid = t >> 5, lane = t & 31;
  const int half = lane >> 4, l16 = lane & 15;
  const int wm = wid & 3, wn = wid >> 2;

  v8f acc[2][4] = {};
  gemm_mainloop(Wbf + (size_t)m0 * 512,
                xnT + ((size_t)b * 1024 + n0) * 512,
                As, Bs, acc, t, wid, lane);

#pragma unroll
  for (int mi = 0; mi < 2; ++mi)
#pragma unroll
    for (int nj = 0; nj < 4; ++nj)
#pragma unroll
      for (int r = 0; r < 8; ++r) {
        int m = m0 + wm * 32 + mi * 16 + r + half * 8;   // o in [0,1536)
        int n = n0 + wn * 64 + nj * 16 + l16;
        float val = acc[mi][nj][r] + bias[m];
        int part = m >> 9, oo = m & 511;
        int h = oo >> 6, dh = oo & 63;
        size_t bh = (size_t)b * 8 + h;
        __bf16 bv = (__bf16)val;
        if (part == 0)      qT[(bh * 1024 + n) * 64 + dh] = bv;
        else if (part == 1) kT[(bh * 1024 + n) * 64 + dh] = bv;
        else                vv[(bh * 64 + dh) * 1024 + n] = bv;
      }
}

// ---------------------------------------------------------------------------
// 3) Flash-style attention: grid (8 q-chunks, 8 heads, 16 batch), 256 thr.
// ---------------------------------------------------------------------------
__global__ __launch_bounds__(256) void attn_kernel(
    const __bf16* __restrict__ qT, const __bf16* __restrict__ kT,
    const __bf16* __restrict__ vv, __bf16* __restrict__ aoT) {
  __shared__ __align__(16) __bf16 Ps[8 * 16 * 32];
  const int t = threadIdx.x, wid = t >> 5, lane = t & 31;
  const int half = lane >> 4, l16 = lane & 15;
  const int h = blockIdx.y, b = blockIdx.z;
  const size_t bh = (size_t)b * 8 + h;
  const int q0 = blockIdx.x * 128 + wid * 16;

  const __bf16* Qb = qT + (bh * 1024 + q0) * 64;
  const __bf16* Kb = kT + bh * 1024 * 64;
  const __bf16* Vb = vv + bh * 64 * 1024;
  __bf16* myP = Ps + wid * 512;

  const v16bf qf0 = frag_a(Qb, 64, lane);
  const v16bf qf1 = frag_a(Qb + 32, 64, lane);

  float mrow[8], lrow[8];
#pragma unroll
  for (int r = 0; r < 8; ++r) { mrow[r] = -1e30f; lrow[r] = 0.f; }
  v8f o[4] = {};

  for (int kb = 0; kb < 32; ++kb) {
    const int m0 = kb * 32;
    v8f s0 = {}, s1 = {};
    v16bf kf;
    kf = frag_b(Kb + (size_t)m0 * 64, 64, lane);             s0 = wmma_bf16(qf0, kf, s0);
    kf = frag_b(Kb + (size_t)m0 * 64 + 32, 64, lane);        s0 = wmma_bf16(qf1, kf, s0);
    kf = frag_b(Kb + (size_t)(m0 + 16) * 64, 64, lane);      s1 = wmma_bf16(qf0, kf, s1);
    kf = frag_b(Kb + (size_t)(m0 + 16) * 64 + 32, 64, lane); s1 = wmma_bf16(qf1, kf, s1);

    float p0[8], p1[8];
#pragma unroll
    for (int r = 0; r < 8; ++r) {
      float a = s0[r] * 0.125f, c = s1[r] * 0.125f;
      float mx = fmaxf(a, c);
#pragma unroll
      for (int d = 8; d > 0; d >>= 1) mx = fmaxf(mx, __shfl_xor(mx, d, 32));
      float mnew = fmaxf(mrow[r], mx);
      float sc = __expf(mrow[r] - mnew);
      float e0 = __expf(a - mnew), e1 = __expf(c - mnew);
      float rs = e0 + e1;
#pragma unroll
      for (int d = 8; d > 0; d >>= 1) rs += __shfl_xor(rs, d, 32);
      lrow[r] = lrow[r] * sc + rs;
      mrow[r] = mnew;
      p0[r] = e0; p1[r] = e1;
#pragma unroll
      for (int u = 0; u < 4; ++u) o[u][r] *= sc;
    }
    // P tile (D layout) -> wave-private LDS as [q][m] for A-fragment reload
#pragma unroll
    for (int r = 0; r < 8; ++r) {
      int q = r + half * 8;
      myP[q * 32 + l16] = (__bf16)p0[r];
      myP[q * 32 + 16 + l16] = (__bf16)p1[r];
    }
    v16bf pf = frag_a(myP, 32, lane);
#pragma unroll
    for (int u = 0; u < 4; ++u) {
      v16bf vf = frag_b(Vb + (size_t)(u * 16) * 1024 + m0, 1024, lane);
      o[u] = wmma_bf16(pf, vf, o[u]);
    }
  }

#pragma unroll
  for (int r = 0; r < 8; ++r) {
    float inv = 1.0f / lrow[r];
    int q = q0 + r + half * 8;
#pragma unroll
    for (int u = 0; u < 4; ++u) {
      int c = h * 64 + u * 16 + l16;
      aoT[((size_t)b * 1024 + q) * 512 + c] = (__bf16)(o[u][r] * inv);
    }
  }
}

// ---------------------------------------------------------------------------
// 4) Proj GEMM + bias + residual (f32 out)
// ---------------------------------------------------------------------------
__global__ __launch_bounds__(256) void proj_gemm(
    const __bf16* __restrict__ Wbf, const float* __restrict__ bias,
    const __bf16* __restrict__ aoT, const float* __restrict__ x,
    float* __restrict__ out) {
  __shared__ __align__(16) __bf16 As[2 * 128 * 32];
  __shared__ __align__(16) __bf16 Bs[2 * 128 * 32];
  const int b = blockIdx.z;
  const int m0 = blockIdx.x * 128, n0 = blockIdx.y * 128;
  const int t = threadIdx.x, wid = t >> 5, lane = t & 31;
  const int half = lane >> 4, l16 = lane & 15;
  const int wm = wid & 3, wn = wid >> 2;

  v8f acc[2][4] = {};
  gemm_mainloop(Wbf + (size_t)m0 * 512,
                aoT + ((size_t)b * 1024 + n0) * 512,
                As, Bs, acc, t, wid, lane);

#pragma unroll
  for (int mi = 0; mi < 2; ++mi)
#pragma unroll
    for (int nj = 0; nj < 4; ++nj)
#pragma unroll
      for (int r = 0; r < 8; ++r) {
        int m = m0 + wm * 32 + mi * 16 + r + half * 8;   // o in [0,512)
        int n = n0 + wn * 64 + nj * 16 + l16;
        size_t idx = ((size_t)b * 512 + m) * 1024 + n;
        __builtin_nontemporal_store(x[idx] + bias[m] + acc[mi][nj][r], &out[idx]);
      }
}

// ---------------------------------------------------------------------------
extern "C" void kernel_launch(void* const* d_in, const int* in_sizes, int n_in,
                              void* d_out, int out_size, void* d_ws, size_t ws_size,
                              hipStream_t stream) {
  const float* x      = (const float*)d_in[0];
  const float* gamma  = (const float*)d_in[1];
  const float* beta   = (const float*)d_in[2];
  const float* w_qkv  = (const float*)d_in[3];
  const float* b_qkv  = (const float*)d_in[4];
  const float* w_proj = (const float*)d_in[5];
  const float* b_proj = (const float*)d_in[6];
  float* out = (float*)d_out;

  const size_t SZ = (size_t)16 * 1024 * 512;   // 8.39M bf16 elems per buffer
  __bf16* ws  = (__bf16*)d_ws;
  __bf16* xnT = ws;
  __bf16* qT  = ws + 1 * SZ;
  __bf16* kT  = ws + 2 * SZ;
  __bf16* vv  = ws + 3 * SZ;
  __bf16* aoT = ws + 4 * SZ;
  __bf16* wqB = ws + 5 * SZ;                   // 1536*512 bf16
  __bf16* wpB = wqB + (size_t)1536 * 512;      // 512*512 bf16

  cvt_kernel<<<dim3(768), 256, 0, stream>>>(w_qkv, wqB);
  cvt_kernel<<<dim3(256), 256, 0, stream>>>(w_proj, wpB);
  gn_kernel<<<dim3(8, 16), 256, 0, stream>>>(x, gamma, beta, xnT);
  qkv_gemm<<<dim3(12, 8, 16), 256, 0, stream>>>(wqB, b_qkv, xnT, qT, kT, vv);
  attn_kernel<<<dim3(8, 8, 16), 256, 0, stream>>>(qT, kT, vv, aoT);
  proj_gemm<<<dim3(4, 8, 16), 256, 0, stream>>>(wpB, b_proj, aoT, x, out);
}
